// GenNeuronStates_11888469475558
// MI455X (gfx1250) — compile-verified, matrix operand order checked by reading, not run on hardware
//
#include <hip/hip_runtime.h>

// ---------------------------------------------------------------------------
// Flash-attention style kernel for gfx1250 (MI455X), wave32 + WMMA bf16.
//   B=2, G=4, H=16, N=1024, dq=dv=64.  Head h lives at stride H=16 in the
//   feature dim.  Output is the contiguous [B,G,H,N,dv] buffer.
// ---------------------------------------------------------------------------

typedef __attribute__((ext_vector_type(16))) __bf16 v16bf;
typedef __attribute__((ext_vector_type(8)))  float  v8f;

#define NB      2
#define NG      4
#define NH      16
#define NSEQ    1024
#define DQ      64
#define DV      64
#define DMODEL  1024          // DQ*NH == DV*NH
#define ROWS_PB 128           // query rows per block (8 waves x 16 rows)
#define KBLK    32            // key columns per inner iteration
#define NITER   (NSEQ / KBLK) // 32

// LDS strides in bf16 elements (row base must stay 16B aligned: *2B % 16 == 0)
#define QSTRIDE 72            // 144 B row
#define KSTRIDE 72
#define VSTRIDE 40            // 80 B row (transposed V: [dv][k])
#define PSTRIDE 40

__device__ __forceinline__ unsigned short f32_to_bf16(float f) {
  union { float f; unsigned u; } x; x.f = f;
  unsigned r = x.u + 0x7FFFu + ((x.u >> 16) & 1u);   // round to nearest even
  return (unsigned short)(r >> 16);
}

union Frag16 { uint4 q[2]; v16bf v; };

// Load a 16-element bf16 fragment per the CDNA5 16-bit A/B VGPR layout:
//   element e (0..7)  holds K = koff + 8*half + e
//   element e (8..15) holds K = koff + 16 + 8*half + (e-8)
// => two contiguous 16-byte LDS loads.
__device__ __forceinline__ v16bf load_frag(const unsigned short* rowbase,
                                           int koff, int half) {
  Frag16 fr;
  fr.q[0] = *(const uint4*)(rowbase + koff + 8 * half);
  fr.q[1] = *(const uint4*)(rowbase + koff + 16 + 8 * half);
  return fr.v;
}

// 16-lane butterfly reductions done entirely in VALU via DPP16 (no LDS).
// quad_perm xor1 (0xB1), quad_perm xor2 (0x4E): each quad holds its own
// reduction; row_half_mirror (0x141, ^7 in 8) combines quads; row_mirror
// (0x140, ^15 in 16) combines the two 8-lane halves.  Never crosses the
// 16-lane group.  CTRL must be a compile-time constant for the builtin.
template <int CTRL>
__device__ __forceinline__ float dpp_perm(float x) {
  return __int_as_float(__builtin_amdgcn_update_dpp(
      0, __float_as_int(x), CTRL, 0xF, 0xF, true));
}
__device__ __forceinline__ float redmax16(float x) {
  x = fmaxf(x, dpp_perm<0xB1>(x));
  x = fmaxf(x, dpp_perm<0x4E>(x));
  x = fmaxf(x, dpp_perm<0x141>(x));
  x = fmaxf(x, dpp_perm<0x140>(x));
  return x;
}
__device__ __forceinline__ float redsum16(float x) {
  x += dpp_perm<0xB1>(x);
  x += dpp_perm<0x4E>(x);
  x += dpp_perm<0x141>(x);
  x += dpp_perm<0x140>(x);
  return x;
}

__global__ __launch_bounds__(256)
void GenNeuronStates_11888469475558_kernel(const float* __restrict__ q,
                                           const float* __restrict__ k,
                                           const float* __restrict__ v,
                                           const float* __restrict__ ncs,
                                           float* __restrict__ out) {
  __shared__ __align__(16) unsigned short q_lds[ROWS_PB * QSTRIDE]; // 18.0 KB
  __shared__ __align__(16) unsigned short k_lds[KBLK   * KSTRIDE];  //  4.5 KB
  __shared__ __align__(16) unsigned short v_lds[DV     * VSTRIDE];  //  5.0 KB
  __shared__ __align__(16) unsigned short p_lds[8 * 16 * PSTRIDE];  // 10.0 KB

  const int tid    = threadIdx.x;
  const int wave   = tid >> 5;
  const int lane   = tid & 31;
  const int lane16 = lane & 15;
  const int half   = lane >> 4;

  // Block decode: g slowest (L2 bias reuse), then (b,h), row-block fastest.
  const int bid     = blockIdx.x;
  const int rowblk  = bid & 7;
  const int head    = bid >> 3;          // 0..127
  const int g       = head >> 5;         // 0..3
  const int b       = (head >> 4) & 1;   // 0..1
  const int h       = head & 15;         // 0..15
  const int rowbase = rowblk * ROWS_PB;

  const size_t bg = (size_t)(b * NG + g);
  const float* qg = q + (bg * NSEQ + rowbase) * DMODEL + h; // [row][d*16]
  const float* kg = k + bg * NSEQ * DMODEL + h;             // [m][d*16]
  const float* vg = v + bg * NSEQ * DMODEL + h;             // [m][n*16]
  const float* biasp = ncs + (size_t)g * NSEQ * NSEQ
                       + (size_t)(rowbase + wave * 16 + 8 * half) * NSEQ
                       + lane16;                            // [t*NSEQ + m]

  // ---- stage Q tile (128 x 64) as bf16 in LDS -----------------------------
  for (int i = tid; i < ROWS_PB * DQ; i += 256) {
    int row = i >> 6, d = i & 63;
    q_lds[row * QSTRIDE + d] = f32_to_bf16(qg[(size_t)row * DMODEL + d * NH]);
  }
  __syncthreads();

  // A fragments of this wave's 16 Q rows (d-chunks 0..31 and 32..63)
  const unsigned short* qrow = &q_lds[(wave * 16 + lane16) * QSTRIDE];
  v16bf aq0 = load_frag(qrow, 0, half);
  v16bf aq1 = load_frag(qrow, 32, half);

  v8f acc0 = {}, acc1 = {}, acc2 = {}, acc3 = {};
  float m_run[8], l_run[8];
#pragma unroll
  for (int t = 0; t < 8; ++t) { m_run[t] = -3.0e38f; l_run[t] = 0.0f; }

  for (int it = 0; it < NITER; ++it) {
    const int m0 = it * KBLK;

    __syncthreads();  // previous iteration's K/V readers are done

    // ---- cooperative bf16 staging of K (row-major d) and V (transposed) ---
    for (int i = tid; i < KBLK * DQ; i += 256) {
      int kr = i >> 6, d = i & 63;
      k_lds[kr * KSTRIDE + d] =
          f32_to_bf16(kg[(size_t)(m0 + kr) * DMODEL + d * NH]);
    }
    for (int i = tid; i < KBLK * DV; i += 256) {
      int kr = i >> 6, n = i & 63;
      v_lds[n * VSTRIDE + kr] =
          f32_to_bf16(vg[(size_t)(m0 + kr) * DMODEL + n * NH]);
    }
    if (m0 + KBLK < NSEQ) {  // prefetch next key/value block into caches
      __builtin_prefetch(&kg[(size_t)(m0 + KBLK + (tid >> 6)) * DMODEL +
                             (tid & 63) * NH], 0, 1);
      __builtin_prefetch(&vg[(size_t)(m0 + KBLK + (tid >> 6)) * DMODEL +
                             (tid & 63) * NH], 0, 1);
    }
    __syncthreads();

    // ---- preload all B fragments so DS latency overlaps WMMA/softmax ------
    const unsigned short* k0 = &k_lds[(0 * 16 + lane16) * KSTRIDE];
    const unsigned short* k1 = &k_lds[(1 * 16 + lane16) * KSTRIDE];
    v16bf bk00 = load_frag(k0, 0, half);
    v16bf bk01 = load_frag(k0, 32, half);
    v16bf bk10 = load_frag(k1, 0, half);
    v16bf bk11 = load_frag(k1, 32, half);
    v16bf bv0 = load_frag(&v_lds[(0 * 16 + lane16) * VSTRIDE], 0, half);
    v16bf bv1 = load_frag(&v_lds[(1 * 16 + lane16) * VSTRIDE], 0, half);
    v16bf bv2 = load_frag(&v_lds[(2 * 16 + lane16) * VSTRIDE], 0, half);
    v16bf bv3 = load_frag(&v_lds[(3 * 16 + lane16) * VSTRIDE], 0, half);

    // ---- S = Q * K^T : two 16x16 column tiles, K-dim 64 = 2 chunks --------
    v8f s0 = {}, s1 = {};
    s0 = __builtin_amdgcn_wmma_f32_16x16x32_bf16(false, aq0, false, bk00,
                                                 (short)0, s0, false, false);
    s0 = __builtin_amdgcn_wmma_f32_16x16x32_bf16(false, aq1, false, bk01,
                                                 (short)0, s0, false, false);
    s1 = __builtin_amdgcn_wmma_f32_16x16x32_bf16(false, aq0, false, bk10,
                                                 (short)0, s1, false, false);
    s1 = __builtin_amdgcn_wmma_f32_16x16x32_bf16(false, aq1, false, bk11,
                                                 (short)0, s1, false, false);

    // ---- bias, online softmax (rows t+8*half live in this 16-lane group) --
#pragma unroll
    for (int t = 0; t < 8; ++t) {
      float x0 = s0[t] * 0.125f - biasp[(size_t)t * NSEQ + m0];
      float x1 = s1[t] * 0.125f - biasp[(size_t)t * NSEQ + m0 + 16];
      float mx   = redmax16(fmaxf(x0, x1));
      float mnew = fmaxf(m_run[t], mx);
      float corr = __expf(m_run[t] - mnew);
      float p0 = __expf(x0 - mnew);
      float p1 = __expf(x1 - mnew);
      float rs = redsum16(p0 + p1);
      l_run[t] = l_run[t] * corr + rs;
      m_run[t] = mnew;
      acc0[t] *= corr; acc1[t] *= corr; acc2[t] *= corr; acc3[t] *= corr;
      // C-layout -> A-layout transpose via per-wave LDS scratch
      int prow = wave * 16 + t + 8 * half;
      p_lds[prow * PSTRIDE + lane16]      = f32_to_bf16(p0);
      p_lds[prow * PSTRIDE + 16 + lane16] = f32_to_bf16(p1);
    }
    // DS ops are in-order within a wave: the load below sees the stores.

    // ---- O += P * V : P is 16x32 (one A frag), V is 32x16 per dv tile -----
    v16bf pfrag =
        load_frag(&p_lds[(wave * 16 + lane16) * PSTRIDE], 0, half);
    acc0 = __builtin_amdgcn_wmma_f32_16x16x32_bf16(false, pfrag, false, bv0,
                                                   (short)0, acc0, false, false);
    acc1 = __builtin_amdgcn_wmma_f32_16x16x32_bf16(false, pfrag, false, bv1,
                                                   (short)0, acc1, false, false);
    acc2 = __builtin_amdgcn_wmma_f32_16x16x32_bf16(false, pfrag, false, bv2,
                                                   (short)0, acc2, false, false);
    acc3 = __builtin_amdgcn_wmma_f32_16x16x32_bf16(false, pfrag, false, bv3,
                                                   (short)0, acc3, false, false);
  }

  // ---- normalize and write out: contiguous [B,G,H,N,dv] -------------------
  const int orow0 = rowbase + wave * 16 + 8 * half;
  float* outp = out + ((size_t)((b * NG + g) * NH + h) * NSEQ + orow0) * DV
                + lane16;
#pragma unroll
  for (int t = 0; t < 8; ++t) {
    float inv = 1.0f / l_run[t];
    outp[(size_t)t * DV + 0]  = acc0[t] * inv;
    outp[(size_t)t * DV + 16] = acc1[t] * inv;
    outp[(size_t)t * DV + 32] = acc2[t] * inv;
    outp[(size_t)t * DV + 48] = acc3[t] * inv;
  }
}

extern "C" void kernel_launch(void* const* d_in, const int* in_sizes, int n_in,
                              void* d_out, int out_size, void* d_ws,
                              size_t ws_size, hipStream_t stream) {
  (void)in_sizes; (void)n_in; (void)out_size; (void)d_ws; (void)ws_size;
  const float* q   = (const float*)d_in[0];
  const float* k   = (const float*)d_in[1];
  const float* v   = (const float*)d_in[2];
  const float* ncs = (const float*)d_in[3];
  float* out = (float*)d_out;

  const int grid = NB * NG * NH * (NSEQ / ROWS_PB); // 1024 blocks
  GenNeuronStates_11888469475558_kernel<<<grid, 256, 0, stream>>>(
      q, k, v, ncs, out);
}
